// BasicSupervisedModel_76733885710651
// MI455X (gfx1250) — compile-verified, matrix operand order checked by reading, not run on hardware
//
#include <hip/hip_runtime.h>
#include <hip/hip_bf16.h>
#include <cfloat>
#include <climits>

#define DIM_D 768
#define DIM_H 1024
#define S_SENT 2000

typedef __attribute__((ext_vector_type(16))) __bf16 v16bf;
typedef __attribute__((ext_vector_type(8)))  float  v8f;

// ---------------- degree / norm prep ----------------

__global__ void k_init(float* __restrict__ deg, float* __restrict__ u,
                       float* __restrict__ rowsum, int n) {
    int i = blockIdx.x * blockDim.x + threadIdx.x;
    if (i < n) { deg[i] = 1.0f; u[i] = 0.0f; rowsum[i] = 0.0f; }  // self-loop weight 1
}

__global__ void k_deg(const int* __restrict__ row, const int* __restrict__ col,
                      const float* __restrict__ ew, float* __restrict__ deg, int e) {
    int i = blockIdx.x * blockDim.x + threadIdx.x;
    if (i < e) atomicAdd(&deg[col[i]], ew[i]);
}

__global__ void k_dinv(const float* __restrict__ deg, float* __restrict__ dinv, int n) {
    int i = blockIdx.x * blockDim.x + threadIdx.x;
    if (i < n) { float d = deg[i]; dinv[i] = (d > 0.0f) ? rsqrtf(d) : 0.0f; }
}

// ---------------- w12 = W1 @ W2  (and c1 = b1 . W2) ----------------

__global__ void k_w12(const float* __restrict__ W1, const float* __restrict__ b1,
                      const float* __restrict__ W2, float* __restrict__ w12,
                      float* __restrict__ c1) {
    __shared__ float red[256];
    int d = blockIdx.x;
    const float* src = (d < DIM_D) ? (W1 + (size_t)d * DIM_H) : b1;
    float acc = 0.0f;
    for (int h = threadIdx.x; h < DIM_H; h += blockDim.x) acc += src[h] * W2[h];
    red[threadIdx.x] = acc;
    __syncthreads();
    for (int s = 128; s > 0; s >>= 1) {
        if ((int)threadIdx.x < s) red[threadIdx.x] += red[threadIdx.x + s];
        __syncthreads();
    }
    if (threadIdx.x == 0) {
        if (d < DIM_D) w12[d] = red[0];
        else           *c1  = red[0];
    }
}

// ---------------- v = x @ w12 via WMMA bf16 (B padded to 16 cols, col 0 live) ----------------

__global__ void __launch_bounds__(128)
k_gemv_wmma(const float* __restrict__ x, const float* __restrict__ w12,
            float* __restrict__ v, int n) {
    const int lane  = threadIdx.x & 31;
    const int wave  = threadIdx.x >> 5;
    const int tile  = blockIdx.x * 4 + wave;
    const int row0  = tile * 16;
    if (row0 >= n) return;                       // wave-uniform: EXEC stays all-ones

    const int  m      = lane & 15;
    const int  khalf  = (lane >= 16) ? 8 : 0;    // A/B 16-bit lane layout: K 0-7,16-23 | 8-15,24-31
    const bool ncol0  = (lane == 0) || (lane == 16);
    int mrow = row0 + m; if (mrow > n - 1) mrow = n - 1;   // tail clamp (harmless dup reads)
    const float* __restrict__ xrow = x + (size_t)mrow * DIM_D;

    v8f c = {};
    for (int k0 = 0; k0 < DIM_D; k0 += 32) {
        __builtin_prefetch(xrow + k0 + 256, 0, 1);         // global_prefetch_b8
        const float4 a0 = *reinterpret_cast<const float4*>(xrow + k0 + khalf);
        const float4 a1 = *reinterpret_cast<const float4*>(xrow + k0 + khalf + 4);
        const float4 a2 = *reinterpret_cast<const float4*>(xrow + k0 + khalf + 16);
        const float4 a3 = *reinterpret_cast<const float4*>(xrow + k0 + khalf + 20);
        v16bf a;
        a[0]=(__bf16)a0.x; a[1]=(__bf16)a0.y; a[2]=(__bf16)a0.z; a[3]=(__bf16)a0.w;
        a[4]=(__bf16)a1.x; a[5]=(__bf16)a1.y; a[6]=(__bf16)a1.z; a[7]=(__bf16)a1.w;
        a[8]=(__bf16)a2.x; a[9]=(__bf16)a2.y; a[10]=(__bf16)a2.z; a[11]=(__bf16)a2.w;
        a[12]=(__bf16)a3.x; a[13]=(__bf16)a3.y; a[14]=(__bf16)a3.z; a[15]=(__bf16)a3.w;

        v16bf b;
        #pragma unroll
        for (int j = 0; j < 8; ++j) {
            b[j]     = ncol0 ? (__bf16)w12[k0 + khalf + j]      : (__bf16)0.0f;
            b[j + 8] = ncol0 ? (__bf16)w12[k0 + khalf + 16 + j] : (__bf16)0.0f;
        }
        c = __builtin_amdgcn_wmma_f32_16x16x32_bf16(false, a, false, b,
                                                    (short)0, c, false, false);
    }
    // C/D layout: VGPR g -> (M=g, N=lane) lanes 0-15 ; (M=8+g, N=lane-16) lanes 16-31.
    // Column N=0 lives in lanes 0 and 16.
    if (lane == 0) {
        #pragma unroll
        for (int g = 0; g < 8; ++g) if (row0 + g < n) v[row0 + g] = c[g];
    } else if (lane == 16) {
        #pragma unroll
        for (int g = 0; g < 8; ++g) if (row0 + 8 + g < n) v[row0 + 8 + g] = c[g];
    }
}

// ---------------- scalar edge scatters ----------------

__global__ void k_scatter1(const int* __restrict__ row, const int* __restrict__ col,
                           const float* __restrict__ ew, const float* __restrict__ dinv,
                           const float* __restrict__ v, float* __restrict__ u,
                           float* __restrict__ rowsum, int e) {
    int i = blockIdx.x * blockDim.x + threadIdx.x;
    if (i < e) {
        int r = row[i], cI = col[i];
        float nm = dinv[r] * ew[i] * dinv[cI];
        atomicAdd(&u[cI], nm * v[r]);
        atomicAdd(&rowsum[cI], nm);
    }
}

// z = A v (add self loop), rs = A 1, att base = self-loop of 2nd conv + const terms
__global__ void k_selfloop(const float* __restrict__ dinv, const float* __restrict__ v,
                           float* __restrict__ u, const float* __restrict__ rowsum,
                           const float* __restrict__ c1, const float* __restrict__ b2,
                           float* __restrict__ att, int n) {
    int i = blockIdx.x * blockDim.x + threadIdx.x;
    if (i < n) {
        float di2 = dinv[i] * dinv[i];
        float z   = u[i] + di2 * v[i];
        u[i] = z;
        float rs = rowsum[i] + di2;
        att[i] = di2 * z + c1[0] * rs + b2[0];
    }
}

__global__ void k_scatter2(const int* __restrict__ row, const int* __restrict__ col,
                           const float* __restrict__ ew, const float* __restrict__ dinv,
                           const float* __restrict__ z, float* __restrict__ att, int e) {
    int i = blockIdx.x * blockDim.x + threadIdx.x;
    if (i < e) {
        int r = row[i], cI = col[i];
        float nm = dinv[r] * ew[i] * dinv[cI];
        atomicAdd(&att[cI], nm * z[r]);
    }
}

// ---------------- top-k cutoff, mask, loss (single workgroup) ----------------

__device__ __forceinline__ float log_sigmoid(float xv) {
    // log_sigmoid(x) = -softplus(-x), stable
    return -(fmaxf(-xv, 0.0f) + log1pf(expf(-fabsf(xv))));
}

__global__ void __launch_bounds__(1024)
k_finish(const float* __restrict__ att, const int* __restrict__ y,
         float* __restrict__ out, int n) {
    __shared__ float s_sent[S_SENT];
    __shared__ float s_t[S_SENT];
    __shared__ unsigned char s_rm[S_SENT];
    __shared__ float red_v[32];
    __shared__ int   red_i[32];
    __shared__ int   s_k;
    __shared__ float s_cut;

    const int tid = threadIdx.x;
    if (tid == 0) s_k = 0;
    for (int j = tid; j < S_SENT; j += blockDim.x) {
        s_sent[j] = att[j]; s_t[j] = 0.0f; s_rm[j] = 0;
    }
    __syncthreads();

    int cnt = 0;
    for (int j = tid; j < S_SENT; j += blockDim.x) if (y[j] >= 0) cnt++;
    atomicAdd(&s_k, cnt);
    for (int j = tid; j < S_SENT; j += blockDim.x) {
        int yy = y[j];
        if (yy > 0 && yy < S_SENT) s_t[yy] = 1.0f;   // scatter-max of 1.0, races benign
    }
    __syncthreads();

    int k = s_k; if (k < 1) k = 1; if (k > S_SENT) k = S_SENT;

    // k-th largest (duplicates counted): remove one occurrence of the max, k times
    for (int it = 0; it < k; ++it) {
        float bv = -FLT_MAX; int bi = INT_MAX;
        for (int j = tid; j < S_SENT; j += blockDim.x) {
            if (!s_rm[j]) {
                float vv = s_sent[j];
                if (vv > bv || (vv == bv && j < bi)) { bv = vv; bi = j; }
            }
        }
        for (int off = 16; off > 0; off >>= 1) {
            float ov = __shfl_down(bv, off, 32);
            int   oi = __shfl_down(bi, off, 32);
            if (ov > bv || (ov == bv && oi < bi)) { bv = ov; bi = oi; }
        }
        int wid = tid >> 5;
        if ((tid & 31) == 0) { red_v[wid] = bv; red_i[wid] = bi; }
        __syncthreads();
        if (tid == 0) {
            float fv = red_v[0]; int fi = red_i[0];
            for (int w = 1; w < (int)(blockDim.x >> 5); ++w)
                if (red_v[w] > fv || (red_v[w] == fv && red_i[w] < fi)) { fv = red_v[w]; fi = red_i[w]; }
            if (fi >= 0 && fi < S_SENT) s_rm[fi] = 1;
            s_cut = fv;
        }
        __syncthreads();
    }
    const float cut = s_cut;

    // loss (deterministic tree reduction)
    float acc = 0.0f;
    for (int j = tid; j < S_SENT; j += blockDim.x) {
        float sj = s_sent[j], tj = s_t[j];
        acc += tj * log_sigmoid(sj) + (1.0f - tj) * log_sigmoid(-sj);
    }
    for (int off = 16; off > 0; off >>= 1) acc += __shfl_down(acc, off, 32);
    if ((tid & 31) == 0) red_v[tid >> 5] = acc;
    __syncthreads();
    if (tid == 0) {
        float s = 0.0f;
        for (int w = 0; w < (int)(blockDim.x >> 5); ++w) s += red_v[w];
        out[0] = -s / (float)S_SENT;
    }
    for (int i = tid; i < n; i += blockDim.x)
        out[1 + i] = (att[i] >= cut) ? 1.0f : 0.0f;
}

// ---------------- host launch ----------------

extern "C" void kernel_launch(void* const* d_in, const int* in_sizes, int n_in,
                              void* d_out, int out_size, void* d_ws, size_t ws_size,
                              hipStream_t stream) {
    const float* x   = (const float*)d_in[0];
    const int*   ei  = (const int*)  d_in[1];
    const float* ew  = (const float*)d_in[2];
    const int*   y   = (const int*)  d_in[3];
    const float* W1  = (const float*)d_in[4];
    const float* b1  = (const float*)d_in[5];
    const float* W2  = (const float*)d_in[6];
    const float* b2  = (const float*)d_in[7];
    float*       out = (float*)d_out;

    const int n = in_sizes[0] / DIM_D;      // 20000
    const int e = in_sizes[2];              // 320000
    const int* row = ei;
    const int* col = ei + e;

    float* ws     = (float*)d_ws;
    float* deg    = ws;
    float* dinv   = ws + (size_t)n;
    float* v      = ws + (size_t)2 * n;
    float* u      = ws + (size_t)3 * n;     // becomes z after k_selfloop
    float* rowsum = ws + (size_t)4 * n;
    float* att    = ws + (size_t)5 * n;
    float* w12    = ws + (size_t)6 * n;
    float* c1     = w12 + DIM_D;

    const int TB = 256;
    k_init   <<<(n + TB - 1) / TB, TB, 0, stream>>>(deg, u, rowsum, n);
    k_deg    <<<(e + TB - 1) / TB, TB, 0, stream>>>(row, col, ew, deg, e);
    k_dinv   <<<(n + TB - 1) / TB, TB, 0, stream>>>(deg, dinv, n);
    k_w12    <<<DIM_D + 1, 256, 0, stream>>>(W1, b1, W2, w12, c1);

    const int tiles = (n + 15) / 16;
    k_gemv_wmma<<<(tiles + 3) / 4, 128, 0, stream>>>(x, w12, v, n);

    k_scatter1<<<(e + TB - 1) / TB, TB, 0, stream>>>(row, col, ew, dinv, v, u, rowsum, e);
    k_selfloop<<<(n + TB - 1) / TB, TB, 0, stream>>>(dinv, v, u, rowsum, c1, b2, att, n);
    k_scatter2<<<(e + TB - 1) / TB, TB, 0, stream>>>(row, col, ew, dinv, u, att, e);

    k_finish<<<1, 1024, 0, stream>>>(att, y, out, n);
}
